// FLlamaAttention_9491877724251
// MI455X (gfx1250) — compile-verified, matrix operand order checked by reading, not run on hardware
//
#include <hip/hip_runtime.h>

#define H    16
#define HKV  4
#define DMODEL 2048
#define HD   128
#define LSEQ 2048
#define REP  (H / HKV)

typedef __attribute__((ext_vector_type(16))) __bf16 v16bf;
typedef __attribute__((ext_vector_type(8)))  __bf16 v8bf;
typedef __attribute__((ext_vector_type(8)))  float  v8f;

// Build a 16x32 bf16 A/B fragment from a row-major bf16 row.
// Per ISA layout: lane half selects K offset; elements 0..7 = K[k0+half*8 .. +8),
// elements 8..15 = K[k0+16+half*8 .. +8). Two 16-byte contiguous loads.
__device__ inline v16bf frag_from_bf16(const __bf16* rowptr, int k0, int half) {
    const v8bf a = *(const v8bf*)(rowptr + k0 + half * 8);
    const v8bf b = *(const v8bf*)(rowptr + k0 + 16 + half * 8);
    v16bf f;
#pragma unroll
    for (int i = 0; i < 8; ++i) { f[i] = a[i]; f[i + 8] = b[i]; }
    return f;
}

// Same fragment, but source is fp32 in memory: load 4x float4, convert to bf16.
__device__ inline v16bf frag_from_f32(const float* rowptr, int k0, int half) {
    const float4 a0 = *(const float4*)(rowptr + k0 + half * 8);
    const float4 a1 = *(const float4*)(rowptr + k0 + half * 8 + 4);
    const float4 b0 = *(const float4*)(rowptr + k0 + half * 8 + 16);
    const float4 b1 = *(const float4*)(rowptr + k0 + half * 8 + 20);
    v16bf f;
    f[0]  = (__bf16)a0.x; f[1]  = (__bf16)a0.y; f[2]  = (__bf16)a0.z; f[3]  = (__bf16)a0.w;
    f[4]  = (__bf16)a1.x; f[5]  = (__bf16)a1.y; f[6]  = (__bf16)a1.z; f[7]  = (__bf16)a1.w;
    f[8]  = (__bf16)b0.x; f[9]  = (__bf16)b0.y; f[10] = (__bf16)b0.z; f[11] = (__bf16)b0.w;
    f[12] = (__bf16)b1.x; f[13] = (__bf16)b1.y; f[14] = (__bf16)b1.z; f[15] = (__bf16)b1.w;
    return f;
}

// ---------------------------------------------------------------------------
// Kernel 1: per-head projection (L x D) @ (D x HD) with optional RoPE, output
// bf16 either row-major [n][l][h] or transposed [n][h][l] (for V).
// Block = 256 threads (8 waves). Wave w owns h-tile w (16 cols). One 16-row
// l-block per block. K-loop over D in chunks of 32 -> 64 WMMAs per wave.
// ---------------------------------------------------------------------------
__global__ void proj_rope_kernel(const float* __restrict__ src,   // (nheads, 1, L, D)
                                 const float* __restrict__ w,     // (nheads, HD, D)
                                 __bf16* __restrict__ dst,
                                 const int* __restrict__ pos_ids, // (1, L)
                                 int do_rope, int transpose_out) {
    __shared__ float tile[16 * HD];
    const int lblk = blockIdx.x;       // 0 .. L/16-1
    const int n    = blockIdx.y;       // head
    const int tid  = threadIdx.x;
    const int wave = tid >> 5;
    const int lane = tid & 31;
    const int half = lane >> 4;
    const int l16  = lane & 15;

    const float* arow = src + ((size_t)n * LSEQ + (size_t)(lblk * 16 + l16)) * DMODEL;
    const float* brow = w + ((size_t)n * HD + (size_t)(wave * 16 + l16)) * DMODEL;

    v8f acc = {};
    for (int k0 = 0; k0 < DMODEL; k0 += 32) {
        const v16bf af = frag_from_f32(arow, k0, half);
        const v16bf bf = frag_from_f32(brow, k0, half);
        acc = __builtin_amdgcn_wmma_f32_16x16x32_bf16(false, af, false, bf,
                                                      (short)0, acc, false, false);
    }
    // Stage 16 x 128 f32 tile in LDS for cross-lane rotate_half.
#pragma unroll
    for (int v = 0; v < 8; ++v)
        tile[(v + half * 8) * HD + wave * 16 + l16] = acc[v];
    __syncthreads();

    for (int e = tid; e < 16 * HD; e += 256) {
        const int r = e >> 7;        // row in tile
        const int c = e & (HD - 1);  // head-dim col
        const int l = lblk * 16 + r;
        float val = tile[r * HD + c];
        if (do_rope) {
            const float p   = (float)pos_ids[l];
            const int   ci  = c & 63;
            // inv_freq = 10000^(-ci/64); ln(10000) = 9.210340371976184
            const float inv = __expf(-(float)ci * (9.210340371976184f / 64.0f));
            const float ang = p * inv;
            const float cs = __cosf(ang), sn = __sinf(ang);
            const float partner = (c < 64) ? -tile[r * HD + c + 64]
                                           :  tile[r * HD + c - 64];
            val = val * cs + partner * sn;
        }
        if (transpose_out)
            dst[((size_t)n * HD + c) * LSEQ + l] = (__bf16)val;
        else
            dst[((size_t)n * LSEQ + l) * HD + c] = (__bf16)val;
    }
}

// ---------------------------------------------------------------------------
// Kernel 2: flash attention. One wave per 16-query tile; resident Q fragments;
// key blocks of 32 (2 sub-tiles of 16); online softmax with shfl_xor row
// reductions over each 16-lane half; P goes through per-wave LDS to switch
// from C-layout to A-fragment layout; o accum = 8 hd-tiles of v8f.
// ---------------------------------------------------------------------------
__global__ void attn_kernel(const __bf16* __restrict__ Q,   // [H][L][HD]
                            const __bf16* __restrict__ K,   // [HKV][L][HD]
                            const __bf16* __restrict__ Vt,  // [HKV][HD][L]
                            __bf16* __restrict__ O) {       // [H][L][HD]
    __shared__ __align__(16) __bf16 plds[8][16 * 32];
    const int tid  = threadIdx.x;
    const int wave = tid >> 5;
    const int lane = tid & 31;
    const int half = lane >> 4;
    const int l16  = lane & 15;
    const int n     = blockIdx.y;
    const int qtile = blockIdx.x * 8 + wave;
    const int qbase = qtile * 16;
    const int kvn   = n / REP;

    const __bf16* qrow = Q + ((size_t)n * LSEQ + qbase + l16) * HD;
    v16bf qf[4];
#pragma unroll
    for (int c = 0; c < 4; ++c) qf[c] = frag_from_bf16(qrow, c * 32, half);

    v8f o[8];
    float m[8], lsum[8];
#pragma unroll
    for (int t = 0; t < 8; ++t) o[t] = {};
#pragma unroll
    for (int v = 0; v < 8; ++v) { m[v] = -3.0e38f; lsum[v] = 0.0f; }

    const float scale = 0.0883883476483184f; // 1/sqrt(128)
    const int kbmax = (qbase + 15) / 32;
    for (int kb = 0; kb <= kbmax; ++kb) {
        // S = Q @ K^T for 32 keys (two 16x16 C tiles)
        v8f sf[2];
#pragma unroll
        for (int sub = 0; sub < 2; ++sub) {
            const __bf16* krow =
                K + ((size_t)kvn * LSEQ + kb * 32 + sub * 16 + l16) * HD;
            v8f s = {};
#pragma unroll
            for (int c = 0; c < 4; ++c) {
                const v16bf kf = frag_from_bf16(krow, c * 32, half);
                s = __builtin_amdgcn_wmma_f32_16x16x32_bf16(false, qf[c], false, kf,
                                                            (short)0, s, false, false);
            }
            sf[sub] = s;
        }
        // scale + causal mask (C layout: row = qbase+v+half*8, col = lane&15)
#pragma unroll
        for (int sub = 0; sub < 2; ++sub) {
            const int col = kb * 32 + sub * 16 + l16;
#pragma unroll
            for (int v = 0; v < 8; ++v) {
                const int row = qbase + v + half * 8;
                const float x = sf[sub][v] * scale;
                sf[sub][v] = (col <= row) ? x : -3.0e38f;
            }
        }
        // online softmax per row (row values live in one 16-lane half)
#pragma unroll
        for (int v = 0; v < 8; ++v) {
            float rm = fmaxf(sf[0][v], sf[1][v]);
#pragma unroll
            for (int off = 8; off >= 1; off >>= 1)
                rm = fmaxf(rm, __shfl_xor(rm, off, 16));
            const float nm   = fmaxf(m[v], rm);
            const float corr = __expf(m[v] - nm);
            const float p0 = __expf(sf[0][v] - nm);
            const float p1 = __expf(sf[1][v] - nm);
            sf[0][v] = p0; sf[1][v] = p1;
            float rs = p0 + p1;
#pragma unroll
            for (int off = 8; off >= 1; off >>= 1)
                rs += __shfl_xor(rs, off, 16);
            lsum[v] = lsum[v] * corr + rs;
            m[v]    = nm;
#pragma unroll
            for (int t = 0; t < 8; ++t) o[t][v] *= corr;
        }
        // C-layout P -> row-major LDS (16x32 bf16) -> A fragment
#pragma unroll
        for (int sub = 0; sub < 2; ++sub)
#pragma unroll
            for (int v = 0; v < 8; ++v)
                plds[wave][(v + half * 8) * 32 + sub * 16 + l16] = (__bf16)sf[sub][v];
        const v16bf pf = frag_from_bf16(&plds[wave][l16 * 32], 0, half);
        // o += P @ V  (V stored transposed: B column = Vt row, contiguous)
#pragma unroll
        for (int t = 0; t < 8; ++t) {
            const __bf16* vrow =
                Vt + ((size_t)kvn * HD + t * 16 + l16) * LSEQ + kb * 32;
            const v16bf vf = frag_from_bf16(vrow, 0, half);
            o[t] = __builtin_amdgcn_wmma_f32_16x16x32_bf16(false, pf, false, vf,
                                                           (short)0, o[t], false, false);
        }
    }
    // normalize + store bf16
#pragma unroll
    for (int t = 0; t < 8; ++t)
#pragma unroll
        for (int v = 0; v < 8; ++v) {
            const float val = o[t][v] / lsum[v];
            O[((size_t)n * LSEQ + qbase + v + half * 8) * HD + t * 16 + l16] = (__bf16)val;
        }
}

// ---------------------------------------------------------------------------
// Kernel 3: per-head output projection (L x HD) @ (HD x D).
// wo is (D, H*HD) row-major: for a fixed output col d the K-run over h is
// contiguous -> plain fp32 loads for the B fragment.
// ---------------------------------------------------------------------------
__global__ void oproj_kernel(const __bf16* __restrict__ Oin,  // [H][L][HD]
                             const float* __restrict__ wo,    // (D, H*HD)
                             float* __restrict__ out) {       // (H, 1, L, D)
    const int tid  = threadIdx.x;
    const int wave = tid >> 5;
    const int lane = tid & 31;
    const int half = lane >> 4;
    const int l16  = lane & 15;
    const int lblk = blockIdx.x;   // 0 .. L/16-1
    const int n    = blockIdx.y;

    const __bf16* arow = Oin + ((size_t)n * LSEQ + lblk * 16 + l16) * HD;
    v16bf af[4];
#pragma unroll
    for (int c = 0; c < 4; ++c) af[c] = frag_from_bf16(arow, c * 32, half);

    for (int t = wave; t < DMODEL / 16; t += 8) {
        const float* brow = wo + (size_t)(t * 16 + l16) * (H * HD) + n * HD;
        v8f acc = {};
#pragma unroll
        for (int c = 0; c < 4; ++c) {
            const v16bf bf = frag_from_f32(brow, c * 32, half);
            acc = __builtin_amdgcn_wmma_f32_16x16x32_bf16(false, af[c], false, bf,
                                                          (short)0, acc, false, false);
        }
#pragma unroll
        for (int v = 0; v < 8; ++v)
            out[((size_t)n * LSEQ + lblk * 16 + v + half * 8) * DMODEL + t * 16 + l16] =
                acc[v];
    }
}

extern "C" void kernel_launch(void* const* d_in, const int* in_sizes, int n_in,
                              void* d_out, int out_size, void* d_ws, size_t ws_size,
                              hipStream_t stream) {
    const float* q_hidden = (const float*)d_in[0];
    const float* k_hidden = (const float*)d_in[1];
    const float* v_hidden = (const float*)d_in[2];
    const float* wq       = (const float*)d_in[3];
    const float* wk       = (const float*)d_in[4];
    const float* wv       = (const float*)d_in[5];
    const float* wo       = (const float*)d_in[6];
    const int*   pos_ids  = (const int*)d_in[7];
    // d_in[8] = attention_mask (pure causal; handled analytically)
    float* out = (float*)d_out;

    // bf16 workspace layout (~20 MB)
    __bf16* Q  = (__bf16*)d_ws;
    __bf16* K  = Q  + (size_t)H   * LSEQ * HD;
    __bf16* Vt = K  + (size_t)HKV * LSEQ * HD;
    __bf16* O  = Vt + (size_t)HKV * HD   * LSEQ;

    const dim3 blk(256);
    proj_rope_kernel<<<dim3(LSEQ / 16, H),   blk, 0, stream>>>(q_hidden, wq, Q,  pos_ids, 1, 0);
    proj_rope_kernel<<<dim3(LSEQ / 16, HKV), blk, 0, stream>>>(k_hidden, wk, K,  pos_ids, 1, 0);
    proj_rope_kernel<<<dim3(LSEQ / 16, HKV), blk, 0, stream>>>(v_hidden, wv, Vt, pos_ids, 0, 1);
    attn_kernel<<<dim3(LSEQ / 16 / 8, H), blk, 0, stream>>>(Q, K, Vt, O);
    oproj_kernel<<<dim3(LSEQ / 16, H), blk, 0, stream>>>(O, wo, out);
}
